// FadeFeatureAlignment_45114336477817
// MI455X (gfx1250) — matrix operand order repaired and checked by reading, not compiled, not run on hardware
//
#include <hip/hip_runtime.h>

typedef __attribute__((ext_vector_type(16))) _Float16 v16h;
typedef __attribute__((ext_vector_type(8)))  _Float16 v8h;
typedef __attribute__((ext_vector_type(4)))  _Float16 v4h;
typedef __attribute__((ext_vector_type(8)))  float    v8f;

// ---------------------------------------------------------------------------
// WMMA GEMM: out[rows x N] = A[rows x 128] * W[128 x N] (+ bias)
// COMBINE epilogue: out = g*val + (1-g)*combineB  (gate at coarse res)
// Block: 256 threads (8 wave32), 128 rows per block. N in {64,128} compile-time
// so every stride folds into immediate offsets and the epilogue is branch-free.
// Fragments are contiguous 16-byte ds_load_b128:
//   A: ISA 16-bit A layout = two contiguous 8-half runs per lane (row-major).
//   B: W pre-swizzled in LDS into fragment-contiguous order at staging.
// ---------------------------------------------------------------------------
template <int N, bool HAS_BIAS, bool COMBINE>
__global__ __launch_bounds__(256)
void gemm_k128(const float* __restrict__ A, const float* __restrict__ W,
               const float* __restrict__ bias, float* __restrict__ out,
               const float* __restrict__ gate, const float* __restrict__ combineB)
{
    constexpr int NTILES = N / 16;
    __shared__ __align__(16) _Float16 Wswz[128 * N];   // fragment-order weights
    __shared__ __align__(16) _Float16 Ah[128 * 128];   // row-major 128-row A tile

    const int tid = threadIdx.x;

    // ---- stage A: float4 global load -> v4h LDS store (row-major f16) ----
    {
        const float4* A4 = (const float4*)(A + (size_t)blockIdx.x * (128 * 128));
#pragma unroll
        for (int i0 = 0; i0 < 16; ++i0) {
            const int i = i0 * 256 + tid;
            float4 x = A4[i];
            union { v4h v; _Float16 h[4]; } q;
            q.h[0] = (_Float16)x.x; q.h[1] = (_Float16)x.y;
            q.h[2] = (_Float16)x.z; q.h[3] = (_Float16)x.w;
            *(v4h*)(Ah + i * 4) = q.v;
        }
    }
    // ---- stage W pre-swizzled: Wswz[((nt*4+ks)*32+lane)*16 + h] ----------
    {
#pragma unroll
        for (int i0 = 0; i0 < NTILES; ++i0) {           // NTILES*256 half-chunks
            const int idx2 = i0 * 256 + tid;
            const int part = idx2 & 1;
            const int frag = idx2 >> 1;
            const int ln   = frag & 31;
            const int ks   = (frag >> 5) & 3;
            const int nt   = frag >> 7;
            const int g    = ln >> 4;
            const int n    = ln & 15;
            union { v8h v; _Float16 h[8]; } q;
#pragma unroll
            for (int j = 0; j < 8; ++j) {
                const int hh = part * 8 + j;
                const int kk = ks * 32 + part * 16 + g * 8 + j;   // B: K = grp*8 + h (+16)
                q.h[j] = (_Float16)W[kk * N + nt * 16 + n];
            }
            *(v8h*)(Wswz + frag * 16 + part * 8) = q.v;
        }
    }
    __syncthreads();

    const int wv   = tid >> 5;      // wave id 0..7 -> 16-row slice
    const int lane = tid & 31;
    const int grp  = lane >> 4;
    const int mn   = lane & 15;     // M for A frag, N for B/C frags

    union AV { v16h v; v8h g[2]; };
    union CV { v8f  v; float f[8]; };

    // A fragments: two contiguous b128 LDS loads per K-step, resident in VGPRs
    AV af[4];
    {
        const _Float16* arow = Ah + (wv * 16 + mn) * 128;
#pragma unroll
        for (int ks = 0; ks < 4; ++ks) {
            const int off = ks * 32 + grp * 8;          // A: K = grp*8 + h (+16)
            af[ks].g[0] = *(const v8h*)(arow + off);
            af[ks].g[1] = *(const v8h*)(arow + off + 16);
        }
    }

    const int rowbase = blockIdx.x * 128 + wv * 16 + grp * 8;

#pragma unroll
    for (int nt = 0; nt < NTILES; ++nt) {
        // batch all B fragment loads, then back-to-back WMMAs
        AV bf[4];
#pragma unroll
        for (int ks = 0; ks < 4; ++ks) {
            const _Float16* bbase = Wswz + ((nt * 4 + ks) * 32 + lane) * 16;
            bf[ks].g[0] = *(const v8h*)(bbase);
            bf[ks].g[1] = *(const v8h*)(bbase + 8);
        }
        v8f cacc = {};
#pragma unroll
        for (int ks = 0; ks < 4; ++ks)
            cacc = __builtin_amdgcn_wmma_f32_16x16x32_f16(
                false, af[ks].v, false, bf[ks].v, (short)0, cacc, false, false);

        CV cu; cu.v = cacc;
        const int   col = nt * 16 + mn;
        const float bv  = HAS_BIAS ? bias[col] : 0.0f;
        float* po = out + (size_t)rowbase * N + col;
#pragma unroll
        for (int v = 0; v < 8; ++v) {
            float val = cu.f[v] + bv;
            if constexpr (COMBINE) {  // pixel decode: B=2, H=W=128
                const int row = rowbase + v;
                const int b = row >> 14, rem = row & 16383;
                const int y = rem >> 7, x = rem & 127;
                const float g = gate[b * 4096 + (y >> 1) * 64 + (x >> 1)];
                val = g * val + (1.0f - g) * combineB[(size_t)row * 128 + col];
            }
            po[v * N] = val;
        }
    }
}

// ---------------------------------------------------------------------------
// Shared 3x3 content conv: emb[.,HH,WW,64] -> logits[.,HH,WW,25], SAME pad.
// Weights (3*3*64*25 = 14400 f32 = 57.6 KB) staged in LDS, broadcast reads.
// ---------------------------------------------------------------------------
__global__ __launch_bounds__(256)
void conv3x3_k25(const float* __restrict__ emb, const float* __restrict__ wc,
                 const float* __restrict__ bc, float* __restrict__ out,
                 int HH, int WW)
{
    __shared__ __align__(16) float Ws[9 * 64 * 25];
    const int tid = threadIdx.x;
    for (int i = tid; i < 3600; i += 256)
        ((float4*)Ws)[i] = ((const float4*)wc)[i];
    __syncthreads();

    const int pix = blockIdx.x * 256 + tid;
    const int b   = pix / (HH * WW);
    const int rem = pix - b * HH * WW;
    const int y   = rem / WW, x = rem - y * WW;

    float acc[25];
#pragma unroll
    for (int k = 0; k < 25; ++k) acc[k] = bc[k];

    for (int dr = -1; dr <= 1; ++dr) {
        const int yy = y + dr;
        if (yy < 0 || yy >= HH) continue;
        for (int dc = -1; dc <= 1; ++dc) {
            const int xx = x + dc;
            if (xx < 0 || xx >= WW) continue;
            const float4* src4 = (const float4*)(emb + ((size_t)(b * HH + yy) * WW + xx) * 64);
            const float*  wp   = Ws + ((dr + 1) * 3 + (dc + 1)) * 64 * 25;
#pragma unroll 4
            for (int e4 = 0; e4 < 16; ++e4) {
                const float4 v = src4[e4];
                const float* wr = wp + e4 * 100;
#pragma unroll
                for (int k = 0; k < 25; ++k)
                    acc[k] += v.x * wr[k] + v.y * wr[25 + k]
                            + v.z * wr[50 + k] + v.w * wr[75 + k];
            }
        }
    }
    float* op = out + (size_t)pix * 25;
#pragma unroll
    for (int k = 0; k < 25; ++k) op[k] = acc[k];
}

// ---------------------------------------------------------------------------
// Gate: sigmoid(coarse . w_gate + b), stored at coarse resolution.
// ---------------------------------------------------------------------------
__global__ __launch_bounds__(256)
void gate_kernel(const float* __restrict__ coarse, const float* __restrict__ wg,
                 const float* __restrict__ bg, float* __restrict__ gate)
{
    const int pix = blockIdx.x * 256 + threadIdx.x;  // < 8192
    const float4* src = (const float4*)(coarse + (size_t)pix * 128);
    const float4* w4  = (const float4*)wg;
    float acc = bg[0];
#pragma unroll 8
    for (int c = 0; c < 32; ++c) {
        const float4 s = src[c], w = w4[c];
        acc += s.x * w.x + s.y * w.y + s.z * w.z + s.w * w.w;
    }
    gate[pix] = 1.0f / (1.0f + __expf(-acc));
}

// ---------------------------------------------------------------------------
// kernel = softmax(f_logits + up2(c_logits)) over 25, written in place.
// ---------------------------------------------------------------------------
__global__ __launch_bounds__(256)
void softmax_kernel(float* __restrict__ flog, const float* __restrict__ clog)
{
    const int pix = blockIdx.x * 256 + threadIdx.x;  // < 32768
    const int b = pix >> 14, rem = pix & 16383;
    const int y = rem >> 7, x = rem & 127;
    const float* cp = clog + (size_t)((b * 64 + (y >> 1)) * 64 + (x >> 1)) * 25;
    float* fp = flog + (size_t)pix * 25;

    float s[25], mx = -3.0e38f;
#pragma unroll
    for (int k = 0; k < 25; ++k) { s[k] = fp[k] + cp[k]; mx = fmaxf(mx, s[k]); }
    float sum = 0.0f;
#pragma unroll
    for (int k = 0; k < 25; ++k) { s[k] = __expf(s[k] - mx); sum += s[k]; }
    const float inv = 1.0f / sum;
#pragma unroll
    for (int k = 0; k < 25; ++k) fp[k] = s[k] * inv;
}

// ---------------------------------------------------------------------------
// CARAFE with the reference's faithful scrambled reshape:
//   carafe[c] = sum_k kernel[k] * patch_flat[c*25 + k],  patch_flat = [pos*128+ch]
// One block per coarse pixel; 5x5x128 patch staged in LDS (float4 loads),
// shared by the four 2x2-upsampled fine pixels.
// ---------------------------------------------------------------------------
__global__ __launch_bounds__(128)
void carafe_kernel(const float* __restrict__ coarse, const float* __restrict__ kern,
                   float* __restrict__ out)
{
    __shared__ __align__(16) float patch[25 * 128];
    __shared__ float kw[4][25];

    const int cb  = blockIdx.x;
    const int b   = cb >> 12, rem = cb & 4095;
    const int hh  = rem >> 6, ww = rem & 63;
    const int tid = threadIdx.x;

    for (int i = tid; i < 800; i += 128) {          // 800 float4 = 25*128 floats
        const int p  = i >> 5;                      // patch position 0..24
        const int c4 = i & 31;                      // channel/4
        const int r = p / 5, cc = p - r * 5;
        const int yy = hh + r - 2, xx = ww + cc - 2;
        float4 v = make_float4(0.f, 0.f, 0.f, 0.f);
        if (yy >= 0 && yy < 64 && xx >= 0 && xx < 64)
            v = ((const float4*)(coarse + (size_t)((b * 64 + yy) * 64 + xx) * 128))[c4];
        ((float4*)patch)[i] = v;
    }
    for (int i = tid; i < 100; i += 128) {
        const int q = i / 25, k = i - q * 25;
        const int fy = hh * 2 + (q >> 1), fx = ww * 2 + (q & 1);
        kw[q][k] = kern[(size_t)((b * 128 + fy) * 128 + fx) * 25 + k];
    }
    __syncthreads();

    const int c = tid;
    const float* pv = patch + c * 25;
#pragma unroll
    for (int q = 0; q < 4; ++q) {
        float acc = 0.0f;
#pragma unroll
        for (int k = 0; k < 25; ++k) acc += kw[q][k] * pv[k];
        const int fy = hh * 2 + (q >> 1), fx = ww * 2 + (q & 1);
        out[(size_t)((b * 128 + fy) * 128 + fx) * 128 + c] = acc;
    }
}

// ---------------------------------------------------------------------------
extern "C" void kernel_launch(void* const* d_in, const int* in_sizes, int n_in,
                              void* d_out, int out_size, void* d_ws, size_t ws_size,
                              hipStream_t stream)
{
    (void)in_sizes; (void)n_in; (void)out_size; (void)ws_size;
    const float* fine      = (const float*)d_in[0];
    const float* coarse    = (const float*)d_in[1];
    const float* w_gate    = (const float*)d_in[2];
    const float* b_gate    = (const float*)d_in[3];
    const float* w_sfine   = (const float*)d_in[4];
    const float* w_scoarse = (const float*)d_in[5];
    const float* b_scoarse = (const float*)d_in[6];
    const float* w_content = (const float*)d_in[7];
    const float* b_content = (const float*)d_in[8];
    const float* w_coarse  = (const float*)d_in[9];
    const float* b_coarse  = (const float*)d_in[10];
    const float* w_fineout = (const float*)d_in[11];
    const float* b_fineout = (const float*)d_in[12];

    float* out = (float*)d_out;
    float* ws  = (float*)d_ws;

    // workspace layout (floats)
    float* f_emb = ws;                                // 32768*64
    float* c_emb = f_emb + 32768 * 64;                //  8192*64
    float* f_log = c_emb + 8192 * 64;                 // 32768*25 (-> softmax in place)
    float* c_log = f_log + 32768 * 25;                //  8192*25
    float* gateb = c_log + 8192 * 25;                 //  8192
    float* car   = gateb + 8192;                      // 32768*128
    float* cout_ = car + (size_t)32768 * 128;         // 32768*128

    // 1) embeddings (WMMA GEMMs, K=128 -> N=64)
    gemm_k128<64, false, false><<<32768 / 128, 256, 0, stream>>>(fine, w_sfine, nullptr, f_emb, nullptr, nullptr);
    gemm_k128<64, true,  false><<< 8192 / 128, 256, 0, stream>>>(coarse, w_scoarse, b_scoarse, c_emb, nullptr, nullptr);

    // 2) shared 3x3 content conv -> kernel logits
    conv3x3_k25<<<32768 / 256, 256, 0, stream>>>(f_emb, w_content, b_content, f_log, 128, 128);
    conv3x3_k25<<< 8192 / 256, 256, 0, stream>>>(c_emb, w_content, b_content, c_log, 64, 64);

    // 3) gate + softmax(kernel)
    gate_kernel<<<8192 / 256, 256, 0, stream>>>(coarse, w_gate, b_gate, gateb);
    softmax_kernel<<<32768 / 256, 256, 0, stream>>>(f_log, c_log);

    // 4) CARAFE reassembly
    carafe_kernel<<<8192, 128, 0, stream>>>(coarse, f_log, car);

    // 5) coarse_out GEMM, then fine_out GEMM with fused gate combine -> d_out
    gemm_k128<128, true, false><<<32768 / 128, 256, 0, stream>>>(car, w_coarse, b_coarse, cout_, nullptr, nullptr);
    gemm_k128<128, true, true ><<<32768 / 128, 256, 0, stream>>>(cout_, w_fineout, b_fineout, out, gateb, cout_);
}
